// HybridGRUQuantum_14310831030380
// MI455X (gfx1250) — matrix operand order repaired and checked by reading, not compiled, not run on hardware
//
#include <hip/hip_runtime.h>

// ---------------------------------------------------------------------------
// HybridGRUQuantum for MI455X (gfx1250, wave32, WMMA, async-to-LDS)
//
// Persistent-kernel GRU scan: 32 workgroups (4 batch-tiles x 8 hidden-tiles),
// recurrent weights resident in LDS (288KB of the 320KB WGP LDS), hidden
// state carried in f16 in global memory so the per-step 64KB h tile is a
// contiguous block staged via GLOBAL_LOAD_ASYNC_TO_LDS_B128 (ASYNCcnt).
// Matrix math: v_wmma_f32_16x16x32_f16. Per-step grid barrier via atomics.
// ---------------------------------------------------------------------------

typedef __attribute__((ext_vector_type(16))) _Float16 v16h;
typedef __attribute__((ext_vector_type(8)))  _Float16 v8h;
typedef __attribute__((ext_vector_type(4)))  _Float16 v4h;
typedef __attribute__((ext_vector_type(8)))  float    v8f;
typedef __attribute__((ext_vector_type(4)))  float    f32x4;

typedef int v4i __attribute__((vector_size(16)));
typedef __attribute__((address_space(1))) v4i gv4i;   // global int4
typedef __attribute__((address_space(3))) v4i lv4i;   // LDS int4
typedef __attribute__((address_space(3))) char las_char;

#define HID    512
#define BATCH  256
#define TSTEPS 1024
#define FEAT   64
#define NBLK   32   // 8 n-tiles * 4 m-tiles

// LDS layout (f16), hot A tiles first so their ds offsets fit 16-bit imm:
//   As  [64][512]  h tile (async-staged every step)          @ 0
//   A2s [64][64]   x tile (cvt-staged every step)            @ 65536
//   B2s [192][64]  input weights [n][k]                      @ 73728
//   Bs  [192][512] recurrent weights (3 gates x 64) [n][k]   @ 98304
#define SMEM_BYTES (64*512*2 + 64*64*2 + 192*64*2 + 192*512*2)  // 294912

#ifndef __has_builtin
#define __has_builtin(x) 0
#endif
#if __has_builtin(__builtin_amdgcn_global_load_async_to_lds_b128)
#define HAVE_ASYNC_COPY_BUILTIN 1
#else
#define HAVE_ASYNC_COPY_BUILTIN 0
#endif
#if __has_builtin(__builtin_amdgcn_s_wait_asynccnt)
#define HAVE_WAIT_ASYNC_BUILTIN 1
#else
#define HAVE_WAIT_ASYNC_BUILTIN 0
#endif

// 16-byte async copy global -> LDS (tracked by ASYNCcnt, no VGPR round trip)
__device__ __forceinline__ void async_copy16(const void* g, void* l) {
#if HAVE_ASYNC_COPY_BUILTIN
  __builtin_amdgcn_global_load_async_to_lds_b128(
      (gv4i*)g, (lv4i*)l, 0, 0);
#else
  unsigned loff = (unsigned)(__UINTPTR_TYPE__)(las_char*)l;
  asm volatile("global_load_async_to_lds_b128 %0, %1, off"
               :: "v"(loff), "v"(g) : "memory");
#endif
}

__device__ __forceinline__ void wait_async0() {
#if HAVE_WAIT_ASYNC_BUILTIN
  __builtin_amdgcn_s_wait_asynccnt(0);
#else
  asm volatile("s_wait_asynccnt 0x0" ::: "memory");
#endif
}

__device__ __forceinline__ v16h frag16(const _Float16* p0, const _Float16* p1) {
  v8h a = *(const v8h*)p0;   // ds_load_b128
  v8h b = *(const v8h*)p1;   // ds_load_b128
  return __builtin_shufflevector(a, b, 0,1,2,3,4,5,6,7,8,9,10,11,12,13,14,15);
}

__device__ __forceinline__ void cvt_store4(_Float16* dst, const float* src) {
  f32x4 v = *(const f32x4*)src;
  v4h h;
  h[0] = (_Float16)v[0]; h[1] = (_Float16)v[1];
  h[2] = (_Float16)v[2]; h[3] = (_Float16)v[3];
  *(v4h*)dst = h;        // ds_store_b64
}

__device__ __forceinline__ float sigmoidf_(float v) {
  return 1.0f / (1.0f + __expf(-v));
}

__global__ void __launch_bounds__(256)
gru_scan_kernel(const float* __restrict__ x,
                const float* __restrict__ w_ih, const float* __restrict__ w_hh,
                const float* __restrict__ b_ih, const float* __restrict__ b_hh,
                _Float16* __restrict__ h0buf, _Float16* __restrict__ h1buf,
                unsigned* __restrict__ barrier_ctr)
{
  extern __shared__ char smem[];
  _Float16* As  = (_Float16*)smem;          // [64][512]
  _Float16* A2s = As  + 64 * 512;           // [64][64]
  _Float16* B2s = A2s + 64 * 64;            // [192][64]
  _Float16* Bs  = B2s + 192 * 64;           // [192][512]

  const int tid    = threadIdx.x;
  const int lane   = tid & 31;
  const int wave   = tid >> 5;
  const int wave_m = wave >> 1;             // 0..3 : 16-row strip within 64
  const int wave_n = wave & 1;              // 0..1 : 32-col strip within 64
  const int laneN  = lane & 15;
  const int laneHi = lane >> 4;             // 0/1
  const int n0 = blockIdx.x * 64;           // hidden-column tile base
  const int m0 = blockIdx.y * 64;           // batch-row tile base

  // ---- One-time: stage weight slices into LDS, transposed to [n][k] f16 ----
  // w_hh is (1536, 512) row-major: row ng is contiguous in k -> coalesced.
  for (int idx = tid; idx < 192 * 128; idx += 256) {
    int n = idx >> 7, k4 = (idx & 127) << 2;
    int g = n >> 6, nl = n & 63;
    int ng = g * 512 + n0 + nl;             // gate column in [0,1536)
    cvt_store4(&Bs[n * 512 + k4], &w_hh[(size_t)ng * 512 + k4]);
  }
  for (int idx = tid; idx < 192 * 16; idx += 256) {
    int n = idx >> 4, k4 = (idx & 15) << 2;
    int g = n >> 6, nl = n & 63;
    int ng = g * 512 + n0 + nl;
    cvt_store4(&B2s[n * 64 + k4], &w_ih[(size_t)ng * 64 + k4]);
  }

  // Hoisted fragment base pointers (laneHi folded in) so all inner-loop
  // ds_load_b128s use pure immediate offsets.
  const _Float16* abase  = &As[(16 * wave_m + laneN) * 512 + laneHi * 16];
  const _Float16* a2base = &A2s[(16 * wave_m + laneN) * 64 + laneHi * 16];
  const _Float16* bbase[2][3];
  const _Float16* b2base[2][3];
  #pragma unroll
  for (int nt = 0; nt < 2; ++nt) {
    const int nl = 32 * wave_n + 16 * nt + laneN;
    #pragma unroll
    for (int g = 0; g < 3; ++g) {
      bbase[nt][g]  = &Bs[(g * 64 + nl) * 512 + laneHi * 16];
      b2base[nt][g] = &B2s[(g * 64 + nl) * 64 + laneHi * 16];
    }
  }
  __syncthreads();

  for (int t = 0; t < TSTEPS; ++t) {
    const _Float16* __restrict__ hin  = (t & 1) ? h1buf : h0buf;
    _Float16*       __restrict__ hout = (t & 1) ? h0buf : h1buf;

    // ---- Stage h tile: contiguous 64KB f16 block, async copy to LDS ----
    {
      const _Float16* gsrc = hin + (size_t)m0 * HID;   // rows m0..m0+63 contiguous
      for (int i = tid; i < 4096; i += 256)            // 4096 x 16B = 64KB
        async_copy16(gsrc + i * 8, As + i * 8);
    }
    // ---- Stage x tile (f32 -> f16 conversion required) ----
    for (int idx = tid; idx < 64 * 16; idx += 256) {
      int m = idx >> 4, k4 = (idx & 15) << 2;
      cvt_store4(&A2s[m * 64 + k4],
                 &x[((size_t)(m0 + m) * TSTEPS + t) * FEAT + k4]);
    }
    if (t + 1 < TSTEPS) {
      // global_prefetch_b8 for next timestep's x tile
      int m = tid & 63;
      __builtin_prefetch(&x[((size_t)(m0 + m) * TSTEPS + (t + 1)) * FEAT], 0, 1);
    }
    wait_async0();        // this wave's async copies landed in LDS
    __syncthreads();      // all waves' copies visible

    const v8f vzero = {0.f,0.f,0.f,0.f,0.f,0.f,0.f,0.f};
    v8f accR[2]  = {vzero, vzero};   // xr + hr
    v8f accZ[2]  = {vzero, vzero};   // xz + hz
    v8f accXN[2] = {vzero, vzero};   // xn only
    v8f accHN[2] = {vzero, vzero};   // hn only

    // ---- Recurrent GEMM: (64x64) x 3 gates, K = 512 ----
    #pragma unroll 4
    for (int ki = 0; ki < 16; ++ki) {
      v16h a = frag16(abase + 32 * ki, abase + 32 * ki + 8);
      #pragma unroll
      for (int nt = 0; nt < 2; ++nt) {
        v16h fbR = frag16(bbase[nt][0] + 32 * ki, bbase[nt][0] + 32 * ki + 8);
        v16h fbZ = frag16(bbase[nt][1] + 32 * ki, bbase[nt][1] + 32 * ki + 8);
        v16h fbN = frag16(bbase[nt][2] + 32 * ki, bbase[nt][2] + 32 * ki + 8);
        accR[nt]  = __builtin_amdgcn_wmma_f32_16x16x32_f16(
            false, a, false, fbR, (short)0, accR[nt],  false, false);
        accZ[nt]  = __builtin_amdgcn_wmma_f32_16x16x32_f16(
            false, a, false, fbZ, (short)0, accZ[nt],  false, false);
        accHN[nt] = __builtin_amdgcn_wmma_f32_16x16x32_f16(
            false, a, false, fbN, (short)0, accHN[nt], false, false);
      }
    }

    // ---- Input GEMM: K = 64 (r/z fold into same accumulators; xn separate) ----
    #pragma unroll
    for (int ki = 0; ki < 2; ++ki) {
      v16h a = frag16(a2base + 32 * ki, a2base + 32 * ki + 8);
      #pragma unroll
      for (int nt = 0; nt < 2; ++nt) {
        v16h fbR = frag16(b2base[nt][0] + 32 * ki, b2base[nt][0] + 32 * ki + 8);
        v16h fbZ = frag16(b2base[nt][1] + 32 * ki, b2base[nt][1] + 32 * ki + 8);
        v16h fbN = frag16(b2base[nt][2] + 32 * ki, b2base[nt][2] + 32 * ki + 8);
        accR[nt]  = __builtin_amdgcn_wmma_f32_16x16x32_f16(
            false, a, false, fbR, (short)0, accR[nt],  false, false);
        accZ[nt]  = __builtin_amdgcn_wmma_f32_16x16x32_f16(
            false, a, false, fbZ, (short)0, accZ[nt],  false, false);
        accXN[nt] = __builtin_amdgcn_wmma_f32_16x16x32_f16(
            false, a, false, fbN, (short)0, accXN[nt], false, false);
      }
    }

    // ---- Gate nonlinearities + h update ----
    // C/D layout: VGPR j -> M = j + 8*laneHi, N = laneN
    #pragma unroll
    for (int nt = 0; nt < 2; ++nt) {
      const int n = n0 + 32 * wave_n + 16 * nt + laneN;
      const float bR  = b_ih[n]        + b_hh[n];
      const float bZ  = b_ih[512 + n]  + b_hh[512 + n];
      const float bXN = b_ih[1024 + n];
      const float bHN = b_hh[1024 + n];
      const int mbase = m0 + 16 * wave_m + 8 * laneHi;
      #pragma unroll
      for (int j = 0; j < 8; ++j) {
        const int m = mbase + j;
        float r  = sigmoidf_(accR[nt][j] + bR);
        float zz = sigmoidf_(accZ[nt][j] + bZ);
        float nn = tanhf(accXN[nt][j] + bXN + r * (accHN[nt][j] + bHN));
        float hp = (float)hin[(size_t)m * HID + n];
        hout[(size_t)m * HID + n] = (_Float16)((1.0f - zz) * nn + zz * hp);
      }
    }

    // ---- Grid-wide barrier: all 32 WGs must finish step t ----
    __threadfence();
    __syncthreads();
    if (tid == 0) {
      atomicAdd(barrier_ctr, 1u);
      const unsigned target = (unsigned)(NBLK) * (unsigned)(t + 1);
      while (*(volatile unsigned*)barrier_ctr < target) {
        __builtin_amdgcn_s_sleep(1);
      }
    }
    __syncthreads();
    __threadfence();
  }
}

__global__ void init_kernel(_Float16* h0, unsigned* ctr) {
  int i = blockIdx.x * blockDim.x + threadIdx.x;
  if (i < BATCH * HID) h0[i] = (_Float16)0.0f;
  if (i == 0) *ctr = 0u;
}

// fc1 + 4-qubit statevector circuit + fc2 + out, one thread per batch element.
__global__ void tail_kernel(const _Float16* __restrict__ hfin,
                            const float* __restrict__ fc1_w, const float* __restrict__ fc1_b,
                            const float* __restrict__ qw,
                            const float* __restrict__ fc2_w, const float* __restrict__ fc2_b,
                            const float* __restrict__ out_w, const float* __restrict__ out_b,
                            float* __restrict__ out)
{
  const int b = blockIdx.x * blockDim.x + threadIdx.x;
  if (b >= BATCH) return;

  // fc1 + tanh -> angles
  float ang[4];
  #pragma unroll
  for (int q = 0; q < 4; ++q) {
    float acc = fc1_b[q];
    for (int k = 0; k < HID; ++k)
      acc += (float)hfin[(size_t)b * HID + k] * fc1_w[q * HID + k];
    ang[q] = tanhf(acc);
  }

  // statevector: qubit q maps to bit (3-q) of the flat index
  float re[16], im[16];
  #pragma unroll
  for (int i = 0; i < 16; ++i) { re[i] = 0.0f; im[i] = 0.0f; }
  re[0] = 1.0f;

  // encoding layer: real rotation [[c,-s],[s,c]] per qubit
  #pragma unroll
  for (int q = 0; q < 4; ++q) {
    const int mask = 8 >> q;
    float s, c;
    __sincosf(0.5f * ang[q], &s, &c);
    #pragma unroll
    for (int i = 0; i < 16; ++i) {
      if (i & mask) continue;
      const int j = i | mask;
      float r0 = re[i], i0 = im[i], r1 = re[j], i1 = im[j];
      re[i] = c * r0 - s * r1;  im[i] = c * i0 - s * i1;
      re[j] = s * r0 + c * r1;  im[j] = s * i0 + c * i1;
    }
  }

  #pragma unroll
  for (int l = 0; l < 2; ++l) {
    #pragma unroll
    for (int q = 0; q < 4; ++q) {
      const float phi = qw[(l * 4 + q) * 3 + 0];
      const float th  = qw[(l * 4 + q) * 3 + 1];
      const float om  = qw[(l * 4 + q) * 3 + 2];
      const float c = __cosf(0.5f * th), s = __sinf(0.5f * th);
      const float ap = 0.5f * (phi + om), am = 0.5f * (phi - om);
      const float cap = __cosf(ap), sap = __sinf(ap);
      const float cam = __cosf(am), sam = __sinf(am);
      // U = [[e^{-i ap} c, -e^{+i am} s], [e^{-i am} s, e^{+i ap} c]]
      const float u00r =  cap * c, u00i = -sap * c;
      const float u01r = -cam * s, u01i = -sam * s;
      const float u10r =  cam * s, u10i = -sam * s;
      const float u11r =  cap * c, u11i =  sap * c;
      const int mask = 8 >> q;
      #pragma unroll
      for (int i = 0; i < 16; ++i) {
        if (i & mask) continue;
        const int j = i | mask;
        float r0 = re[i], i0 = im[i], r1 = re[j], i1 = im[j];
        re[i] = u00r * r0 - u00i * i0 + u01r * r1 - u01i * i1;
        im[i] = u00r * i0 + u00i * r0 + u01r * i1 + u01i * r1;
        re[j] = u10r * r0 - u10i * i0 + u11r * r1 - u11i * i1;
        im[j] = u10r * i0 + u10i * r0 + u11r * i1 + u11i * r1;
      }
    }
    const int rr = l % 3 + 1;
    #pragma unroll
    for (int q = 0; q < 4; ++q) {
      const int mc = 8 >> q, mt = 8 >> ((q + rr) & 3);
      #pragma unroll
      for (int i = 0; i < 16; ++i) {
        if ((i & mc) && !(i & mt)) {
          const int j = i | mt;
          float tr = re[i], ti = im[i];
          re[i] = re[j]; im[i] = im[j];
          re[j] = tr;    im[j] = ti;
        }
      }
    }
  }

  // <Z_q> expectation values
  float z[4];
  #pragma unroll
  for (int q = 0; q < 4; ++q) {
    const int mask = 8 >> q;
    float acc = 0.0f;
    #pragma unroll
    for (int i = 0; i < 16; ++i) {
      const float p = re[i] * re[i] + im[i] * im[i];
      acc += (i & mask) ? -p : p;
    }
    z[q] = acc;
  }

  // fc2 + relu, then out
  float o = out_b[0];
  #pragma unroll
  for (int oi = 0; oi < 16; ++oi) {
    float a = fc2_b[oi];
    #pragma unroll
    for (int q = 0; q < 4; ++q) a += z[q] * fc2_w[oi * 4 + q];
    a = fmaxf(a, 0.0f);
    o += a * out_w[oi];
  }
  out[b] = o;
}

extern "C" void kernel_launch(void* const* d_in, const int* in_sizes, int n_in,
                              void* d_out, int out_size, void* d_ws, size_t ws_size,
                              hipStream_t stream) {
  (void)in_sizes; (void)n_in; (void)out_size; (void)ws_size;

  const float* x     = (const float*)d_in[0];
  const float* w_ih  = (const float*)d_in[1];
  const float* w_hh  = (const float*)d_in[2];
  const float* b_ih  = (const float*)d_in[3];
  const float* b_hh  = (const float*)d_in[4];
  const float* fc1_w = (const float*)d_in[5];
  const float* fc1_b = (const float*)d_in[6];
  const float* qw    = (const float*)d_in[7];
  const float* fc2_w = (const float*)d_in[8];
  const float* fc2_b = (const float*)d_in[9];
  const float* out_w = (const float*)d_in[10];
  const float* out_b = (const float*)d_in[11];
  float* out = (float*)d_out;

  // workspace: two f16 h ping-pong buffers + barrier counter
  _Float16* h0 = (_Float16*)d_ws;
  _Float16* h1 = h0 + (size_t)BATCH * HID;
  unsigned* ctr = (unsigned*)(h1 + (size_t)BATCH * HID);

  // allow 288KB dynamic LDS (WGP supports 320KB on gfx1250)
  (void)hipFuncSetAttribute(reinterpret_cast<const void*>(gru_scan_kernel),
                            hipFuncAttributeMaxDynamicSharedMemorySize, SMEM_BYTES);

  init_kernel<<<(BATCH * HID + 255) / 256, 256, 0, stream>>>(h0, ctr);

  gru_scan_kernel<<<dim3(8, 4), 256, SMEM_BYTES, stream>>>(
      x, w_ih, w_hh, b_ih, b_hh, h0, h1, ctr);

  // final hidden state ends up in h0 (1024 steps, even)
  tail_kernel<<<(BATCH + 63) / 64, 64, 0, stream>>>(
      h0, fc1_w, fc1_b, qw, fc2_w, fc2_b, out_w, out_b, out);
}
